// IndRNN_45535243272431
// MI455X (gfx1250) — matrix operand order repaired and checked by reading, not compiled, not run on hardware
//
#include <hip/hip_runtime.h>

// IndRNN on MI455X (gfx1250):
//   Phase 0: convert x [T*B, I] and w_ih [H, I] f32 -> bf16 (RNE) into d_ws.
//   Phase 1: pre = Xb @ Wb^T + bias  via v_wmma_f32_16x16x32_bf16, f32 accum,
//            written into d_out's ys region.  Block tile 256M x 64N, 8 waves,
//            wave tile 32M x 64N (8 accumulators), K stepped by 32.
//            All 4 B-fragments are loaded into distinct buffers before the
//            WMMA burst so loads stay in flight across the whole K-step.
//   Phase 2: sequential scan over T in place, software-pipelined (32-deep
//            register buffer, double buffered) to hide HBM latency.

typedef __attribute__((ext_vector_type(16))) __bf16 bf16x16;
typedef __attribute__((ext_vector_type(8)))  float  f32x8;

union BFrag {
    uint4    q[2];
    unsigned u[8];
    bf16x16  v;
};

// f32 -> bf16 round-to-nearest-even, packed pair into one dword.
__device__ __forceinline__ unsigned pk_bf16(float lo, float hi) {
    unsigned ul = __builtin_bit_cast(unsigned, lo);
    unsigned uh = __builtin_bit_cast(unsigned, hi);
    ul = (ul + 0x7FFFu + ((ul >> 16) & 1u)) >> 16;
    uh = (uh + 0x7FFFu + ((uh >> 16) & 1u)) >> 16;
    return ul | (uh << 16);
}

// ---------------------------------------------------------------------------
// Phase 0: bulk f32 -> bf16 conversion, 8 elements / thread (32B in, 16B out)
// ---------------------------------------------------------------------------
__global__ __launch_bounds__(256) void f32_to_bf16_vec8(
    const float* __restrict__ in, uint4* __restrict__ out, int nvec8)
{
    const int t = blockIdx.x * 256 + threadIdx.x;
    if (t >= nvec8) return;
    const float4* p = (const float4*)in + (size_t)t * 2;
    const float4 a = p[0];
    const float4 b = p[1];
    uint4 o;
    o.x = pk_bf16(a.x, a.y);
    o.y = pk_bf16(a.z, a.w);
    o.z = pk_bf16(b.x, b.y);
    o.w = pk_bf16(b.z, b.w);
    out[t] = o;
}

// ---------------------------------------------------------------------------
// Phase 1: GEMM  Pre[M,N] = Xb[M,K] * Wb[N,K]^T + bias
//   M = 65536, N = 1024, K = 1024, bf16 inputs, f32 accumulate.
// ---------------------------------------------------------------------------
__global__ __launch_bounds__(256) void indrnn_gemm_bf16(
    const unsigned short* __restrict__ Xb,   // [M,K] bf16
    const unsigned short* __restrict__ Wb,   // [N,K] bf16
    const float* __restrict__ Bih,
    const float* __restrict__ Bhh,
    float* __restrict__ Pre,
    int M, int N, int K)
{
    const int lane = threadIdx.x & 31;
    const int wave = threadIdx.x >> 5;

    const int n0 = blockIdx.x * 64;          // 16 N-tiles (fastest varying)
    const int m0 = blockIdx.y * 256 + wave * 32;

    // A-fragment (16x32 bf16): lanes 0-15 row=+lane, K {0..15};
    //                          lanes 16-31 row=+lane-16, K {16..31}.
    const int aRowL = (lane & 15);
    const int aKoff = (lane >> 4) * 16;
    // B-fragment (32x16 bf16): lane&15 = column N, lane>>4 = K-half (0/16).
    const int bColL = (lane & 15);
    const int bKoff = (lane >> 4) * 16;

    const unsigned short* __restrict__ xr0 =
        Xb + (size_t)(m0 + aRowL) * (size_t)K + aKoff;
    const unsigned short* __restrict__ xr1 = xr0 + (size_t)16 * (size_t)K;

    // Per-column B row pointers (K-contiguous in memory).
    const unsigned short* __restrict__ wr0 =
        Wb + (size_t)(n0 + 0 * 16 + bColL) * (size_t)K + bKoff;
    const unsigned short* __restrict__ wr1 = wr0 + (size_t)16 * (size_t)K;
    const unsigned short* __restrict__ wr2 = wr0 + (size_t)32 * (size_t)K;
    const unsigned short* __restrict__ wr3 = wr0 + (size_t)48 * (size_t)K;

    f32x8 acc[2][4] = {};

    for (int k0 = 0; k0 < K; k0 += 32) {
        // ---- issue all 12 loads for this K-step into distinct buffers ----
        BFrag a0, a1, b0, b1, b2, b3;
        a0.q[0] = *(const uint4*)(xr0 + k0);
        a0.q[1] = *(const uint4*)(xr0 + k0 + 8);
        a1.q[0] = *(const uint4*)(xr1 + k0);
        a1.q[1] = *(const uint4*)(xr1 + k0 + 8);
        b0.q[0] = *(const uint4*)(wr0 + k0);
        b0.q[1] = *(const uint4*)(wr0 + k0 + 8);
        b1.q[0] = *(const uint4*)(wr1 + k0);
        b1.q[1] = *(const uint4*)(wr1 + k0 + 8);
        b2.q[0] = *(const uint4*)(wr2 + k0);
        b2.q[1] = *(const uint4*)(wr2 + k0 + 8);
        b3.q[0] = *(const uint4*)(wr3 + k0);
        b3.q[1] = *(const uint4*)(wr3 + k0 + 8);

        // ---- 8 WMMAs; waits can decay monotonically across the burst ----
        acc[0][0] = __builtin_amdgcn_wmma_f32_16x16x32_bf16(
            false, a0.v, false, b0.v, (short)0, acc[0][0], false, false);
        acc[1][0] = __builtin_amdgcn_wmma_f32_16x16x32_bf16(
            false, a1.v, false, b0.v, (short)0, acc[1][0], false, false);
        acc[0][1] = __builtin_amdgcn_wmma_f32_16x16x32_bf16(
            false, a0.v, false, b1.v, (short)0, acc[0][1], false, false);
        acc[1][1] = __builtin_amdgcn_wmma_f32_16x16x32_bf16(
            false, a1.v, false, b1.v, (short)0, acc[1][1], false, false);
        acc[0][2] = __builtin_amdgcn_wmma_f32_16x16x32_bf16(
            false, a0.v, false, b2.v, (short)0, acc[0][2], false, false);
        acc[1][2] = __builtin_amdgcn_wmma_f32_16x16x32_bf16(
            false, a1.v, false, b2.v, (short)0, acc[1][2], false, false);
        acc[0][3] = __builtin_amdgcn_wmma_f32_16x16x32_bf16(
            false, a0.v, false, b3.v, (short)0, acc[0][3], false, false);
        acc[1][3] = __builtin_amdgcn_wmma_f32_16x16x32_bf16(
            false, a1.v, false, b3.v, (short)0, acc[1][3], false, false);
    }

    // C layout: VGPR r -> row = base + r + 8*(lane>>4), col = n0+j*16+(lane&15)
    #pragma unroll
    for (int g = 0; g < 2; ++g) {
        const int rowBase = m0 + g * 16 + ((lane >> 4) * 8);
        #pragma unroll
        for (int j = 0; j < 4; ++j) {
            const int col = n0 + j * 16 + (lane & 15);
            const float bias = Bih[col] + Bhh[col];
            float* __restrict__ o = Pre + (size_t)rowBase * (size_t)N + col;
            #pragma unroll
            for (int r = 0; r < 8; ++r)
                o[(size_t)r * (size_t)N] = acc[g][j][r] + bias;
        }
    }
}

// ---------------------------------------------------------------------------
// Phase 2: in-place scan, one thread per (b,h) chain, 32-deep double-buffered
// software pipeline: loads of block k+1 issue before compute/store of block k.
// ---------------------------------------------------------------------------
__global__ __launch_bounds__(256) void indrnn_scan(
    float* __restrict__ Y,          // [T,B,H], in place (pre -> ys)
    const float* __restrict__ U,    // [H]
    float* __restrict__ Hlast,      // [B,H]
    int T, int B, int H)
{
    const int tid = blockIdx.x * 256 + threadIdx.x;  // b*H + h
    const int h = tid & (H - 1);
    const float uu = U[h];
    const size_t stride = (size_t)B * (size_t)H;     // elements between timesteps

    constexpr int UF = 32;                           // pipeline depth
    float bufA[UF], bufB[UF];
    float hs = 0.0f;

    const float* pl = Y + tid;                       // load cursor
    float*       ps = Y + tid;                       // store cursor

    // prologue: load block 0
    #pragma unroll
    for (int i = 0; i < UF; ++i) bufA[i] = pl[(size_t)i * stride];
    pl += (size_t)UF * stride;

    const int NB = T / UF;                           // 64 blocks
    for (int pr = 0; pr < (NB - 2) / 2; ++pr) {      // 31 double-block iters
        #pragma unroll
        for (int i = 0; i < UF; ++i) bufB[i] = pl[(size_t)i * stride];
        pl += (size_t)UF * stride;
        #pragma unroll
        for (int i = 0; i < UF; ++i) {
            float v = bufA[i] + uu * hs;
            hs = v > 0.0f ? v : 0.0f;
            ps[(size_t)i * stride] = hs;
        }
        ps += (size_t)UF * stride;

        #pragma unroll
        for (int i = 0; i < UF; ++i) bufA[i] = pl[(size_t)i * stride];
        pl += (size_t)UF * stride;
        #pragma unroll
        for (int i = 0; i < UF; ++i) {
            float v = bufB[i] + uu * hs;
            hs = v > 0.0f ? v : 0.0f;
            ps[(size_t)i * stride] = hs;
        }
        ps += (size_t)UF * stride;
    }

    // epilogue: bufA holds block NB-2; load and run final two blocks
    #pragma unroll
    for (int i = 0; i < UF; ++i) bufB[i] = pl[(size_t)i * stride];
    #pragma unroll
    for (int i = 0; i < UF; ++i) {
        float v = bufA[i] + uu * hs;
        hs = v > 0.0f ? v : 0.0f;
        ps[(size_t)i * stride] = hs;
    }
    ps += (size_t)UF * stride;
    #pragma unroll
    for (int i = 0; i < UF; ++i) {
        float v = bufB[i] + uu * hs;
        hs = v > 0.0f ? v : 0.0f;
        ps[(size_t)i * stride] = hs;
    }
    Hlast[tid] = hs;
}

extern "C" void kernel_launch(void* const* d_in, const int* in_sizes, int n_in,
                              void* d_out, int out_size, void* d_ws, size_t ws_size,
                              hipStream_t stream) {
    (void)in_sizes; (void)n_in; (void)out_size; (void)ws_size;

    const float* x    = (const float*)d_in[0];   // [T,B,I]
    const float* w_ih = (const float*)d_in[1];   // [H,I]
    const float* b_ih = (const float*)d_in[2];   // [H]
    const float* b_hh = (const float*)d_in[3];   // [H]
    const float* u    = (const float*)d_in[4];   // [H]
    float* out = (float*)d_out;                  // ys [T,B,H] then h_last [B,H]

    constexpr int T = 2048, B = 32, H = 1024, I = 1024;
    constexpr int M = T * B;                     // 65536

    // Workspace: bf16 copies of x (128 MB) and w (2 MB)
    unsigned short* xb = (unsigned short*)d_ws;
    unsigned short* wb = xb + (size_t)M * (size_t)I;

    // Phase 0: conversions
    {
        const int nx8 = (M * (I / 8));           // 8,388,608 groups of 8
        f32_to_bf16_vec8<<<nx8 / 256, 256, 0, stream>>>(x, (uint4*)xb, nx8);
        const int nw8 = (H * (I / 8));           // 131,072 groups of 8
        f32_to_bf16_vec8<<<nw8 / 256, 256, 0, stream>>>(w_ih, (uint4*)wb, nw8);
    }

    // Phase 1: GEMM -> pre into ys region of d_out
    {
        dim3 grid(H / 64, M / 256);
        indrnn_gemm_bf16<<<grid, 256, 0, stream>>>(xb, wb, b_ih, b_hh, out, M, H, I);
    }

    // Phase 2: in-place scan; h_last appended after ys
    {
        float* hlast = out + (size_t)M * (size_t)H;
        indrnn_scan<<<(B * H) / 256, 256, 0, stream>>>(out, u, hlast, T, B, H);
    }
}
